// Generator_68161130988137
// MI455X (gfx1250) — matrix operand order repaired and checked by reading, not compiled, not run on hardware
//
#include <hip/hip_runtime.h>

// ---- problem constants (from setup_inputs) ----
#define BATCH 16
#define SEQ   2382
#define NPAD  2400              // SEQ padded to multiple of 32
#define NFULL 2368              // largest multiple of 32 <= SEQ (74 unmasked chunks)
#define NTOT  (BATCH*SEQ)       // 38112
#define EPSBN 1e-5f
#define LOG2E 1.44269504088896340736f
#define TILES ((SEQ+15)/16)     // 149 n-tiles of 16
#define TPB   ((TILES+3)/4)     // 38 blocks/batch, 4 waves each

typedef __attribute__((ext_vector_type(16))) _Float16 v16h;
typedef __attribute__((ext_vector_type(8)))  float    v8f;
typedef __attribute__((ext_vector_type(8)))  float    v8ff;

// params layout (floats):
// [0..7] A_o   [8..15] C'_o   [16..23] sn_wq   [24..31] sn_wk
// [32] bq  [33] bk  [34] b2   [40..47] bv      [48..111] sn_wv[h*8+c]
// [112..119] sn_w2
__global__ void gen_k1_stats(const float* __restrict__ z,
                             const float* __restrict__ w1, const float* __restrict__ b1,
                             const float* __restrict__ g1, const float* __restrict__ be1,
                             const float* __restrict__ wq, const float* __restrict__ bq,
                             const float* __restrict__ wk, const float* __restrict__ bk,
                             const float* __restrict__ wv, const float* __restrict__ bv,
                             const float* __restrict__ w2, const float* __restrict__ b2,
                             float* __restrict__ params) {
  __shared__ float ss[256], ss2[256];
  int t = threadIdx.x;
  float s = 0.f, s2 = 0.f;
  for (int i = t; i < NTOT; i += 256) { float v = z[i]; s += v; s2 += v * v; }
  ss[t] = s; ss2[t] = s2; __syncthreads();
  for (int k = 128; k > 0; k >>= 1) {
    if (t < k) { ss[t] += ss[t + k]; ss2[t] += ss2[t + k]; }
    __syncthreads();
  }
  if (t == 0) {
    float mz = ss[0] / (float)NTOT;
    float vz = ss2[0] / (float)NTOT - mz * mz;
    // rank-1 spectral norms = vector 2-norms
    float nw1 = 0.f, nwq = 0.f, nwk = 0.f, nw2 = 0.f;
    for (int o = 0; o < 8; ++o) {
      nw1 += w1[o] * w1[o]; nwq += wq[o] * wq[o];
      nwk += wk[o] * wk[o]; nw2 += w2[o] * w2[o];
    }
    nw1 = sqrtf(nw1); nwq = sqrtf(nwq); nwk = sqrtf(nwk); nw2 = sqrtf(nw2);
    // wv (8x8): power iteration on G = wv^T wv
    float G[8][8];
    for (int i = 0; i < 8; ++i)
      for (int j = 0; j < 8; ++j) {
        float a = 0.f;
        for (int r = 0; r < 8; ++r) a += wv[r * 8 + i] * wv[r * 8 + j];
        G[i][j] = a;
      }
    float v[8]; for (int i = 0; i < 8; ++i) v[i] = 1.0f;
    for (int it = 0; it < 200; ++it) {
      float u[8]; float nn = 0.f;
      for (int i = 0; i < 8; ++i) {
        float a = 0.f;
        for (int j = 0; j < 8; ++j) a += G[i][j] * v[j];
        u[i] = a; nn += a * a;
      }
      nn = rsqrtf(nn + 1e-30f);
      for (int i = 0; i < 8; ++i) v[i] = u[i] * nn;
    }
    float lam = 0.f;
    for (int i = 0; i < 8; ++i) {
      float a = 0.f;
      for (int j = 0; j < 8; ++j) a += G[i][j] * v[j];
      lam += v[i] * a;
    }
    float sigv = sqrtf(fmaxf(lam, 1e-30f));
    for (int o = 0; o < 8; ++o) {
      float sw1 = w1[o] / nw1;
      float A = sw1 * g1[o] * rsqrtf(sw1 * sw1 * vz + EPSBN);
      params[o]      = A;
      params[8 + o]  = be1[o] - A * mz;   // fold BN shift
      params[16 + o] = wq[o] / nwq;
      params[24 + o] = wk[o] / nwk;
      params[40 + o] = bv[o];
      params[112 + o] = w2[o] / nw2;
      for (int c = 0; c < 8; ++c) params[48 + o * 8 + c] = wv[o * 8 + c] / sigv;
    }
    params[32] = bq[0]; params[33] = bk[0]; params[34] = b2[0];
  }
}

// x (f32 [B][8][SEQ]), q (f32 [B][SEQ]), k (f32 [B][NPAD], zero-padded),
// vpad (f16 [B][16][NPAD]): rows 0-7 = V, row 8 = 1.0 (valid m), rows 9-15 = 0
__global__ void gen_k2_features(const float* __restrict__ z,
                                const float* __restrict__ params,
                                float* __restrict__ x, float* __restrict__ q,
                                float* __restrict__ k, _Float16* __restrict__ vpad) {
  int idx = blockIdx.x * blockDim.x + threadIdx.x;   // covers B*NPAD exactly
  int b = idx / NPAD, n = idx % NPAD;
  if (n >= SEQ) {
    k[b * NPAD + n] = 0.f;
    #pragma unroll
    for (int h = 0; h < 16; ++h) vpad[(b * 16 + h) * NPAD + n] = (_Float16)0.f;
    return;
  }
  float zv = z[b * SEQ + n];
  float xo[8];
  #pragma unroll
  for (int o = 0; o < 8; ++o) xo[o] = fmaxf(params[o] * zv + params[8 + o], 0.f);
  float qq = params[32], kk = params[33];
  #pragma unroll
  for (int o = 0; o < 8; ++o) { qq += params[16 + o] * xo[o]; kk += params[24 + o] * xo[o]; }
  q[b * SEQ + n] = qq;
  k[b * NPAD + n] = kk;
  #pragma unroll
  for (int o = 0; o < 8; ++o) x[(b * 8 + o) * SEQ + n] = xo[o];
  #pragma unroll
  for (int h = 0; h < 8; ++h) {
    float vv = params[40 + h];
    #pragma unroll
    for (int c = 0; c < 8; ++c) vv += params[48 + h * 8 + c] * xo[c];
    vpad[(b * 16 + h) * NPAD + n] = (_Float16)vv;
  }
  vpad[(b * 16 + 8) * NPAD + n] = (_Float16)1.0f;
  #pragma unroll
  for (int h = 9; h < 16; ++h) vpad[(b * 16 + h) * NPAD + n] = (_Float16)0.f;
}

__global__ void gen_k2b_kstat(const float* __restrict__ k, float* __restrict__ kstat) {
  __shared__ float smax[256], smin[256];
  int b = blockIdx.x, t = threadIdx.x;
  float mx = -1e30f, mn = 1e30f;
  for (int n = t; n < SEQ; n += 256) {
    float v = k[b * NPAD + n];
    mx = fmaxf(mx, v); mn = fminf(mn, v);
  }
  smax[t] = mx; smin[t] = mn; __syncthreads();
  for (int s = 128; s > 0; s >>= 1) {
    if (t < s) { smax[t] = fmaxf(smax[t], smax[t + s]); smin[t] = fminf(smin[t], smin[t + s]); }
    __syncthreads();
  }
  if (t == 0) { kstat[b * 2] = smax[0]; kstat[b * 2 + 1] = smin[0]; }
}

// one wave per 16-row n-tile; stream m in chunks of 32 through v_wmma_f32_16x16x32_f16
__global__ void gen_k3_attn(const float* __restrict__ q, const float* __restrict__ k,
                            const _Float16* __restrict__ vpad, const float* __restrict__ x,
                            const float* __restrict__ kstat, const float* __restrict__ params,
                            const float* __restrict__ gamma_p, float* __restrict__ ypre) {
  int wave = threadIdx.x >> 5;
  int tile = (blockIdx.x % TPB) * 4 + wave;
  int b = blockIdx.x / TPB;
  if (tile >= TILES) return;                 // whole-wave uniform exit (EXEC stays all-1 for WMMA)
  int lane = threadIdx.x & 31;
  int hi = lane >> 4;                        // half-wave select
  int lm = lane & 15;                        // A: row M ; B/D: column (h)
  int n0 = tile * 16;
  float gamma = gamma_p[0];
  float kmax = kstat[b * 2], kmin = kstat[b * 2 + 1];
  int nq = n0 + lm;
  float qM = (nq < SEQ) ? q[b * SEQ + nq] : 0.f;
  float rowmax = (qM >= 0.f) ? qM * kmax : qM * kmin;  // max_m q*k -> E in [0,1], f16-safe
  float ql2 = qM * LOG2E;                    // exp(q*k - rm) = exp2(ql2*k - rm2): 1 fma + 1 v_exp
  float rm2 = rowmax * LOG2E;

  v8f acc = {};
  const float* kb = k + b * NPAD;
  const _Float16* vb = vpad + (b * 16 + lm) * NPAD + hi * 16;

  // ---- main loop: 74 fully-valid chunks, no per-element masking ----
  #pragma unroll 2
  for (int mc = 0; mc < NFULL; mc += 32) {
    // A tile: E[M][K], contiguous k order, same as B: lane holds 16 consecutive K at hi*16 (K = hi*16 + j)
    v8ff ka = *(const v8ff*)(kb + mc + hi * 16);
    v8ff kc = *(const v8ff*)(kb + mc + hi * 16 + 8);
    v16h A;
    #pragma unroll
    for (int j = 0; j < 16; ++j) {
      float kvj = (j < 8) ? ka[j] : kc[j - 8];
      A[j] = (_Float16)__builtin_amdgcn_exp2f(ql2 * kvj - rm2);
    }
    // B tile: Vpad[K][h], contiguous k order, same as A: lane holds 16 consecutive K at hi*16
    v16h B = *(const v16h*)(vb + mc);
    acc = __builtin_amdgcn_wmma_f32_16x16x32_f16(false, A, false, B,
                                                 (short)0, acc, false, false);
  }
  // ---- tail chunk (mc = NFULL): mask columns m >= SEQ (avoid 0*Inf NaN) ----
  {
    const int mc = NFULL;
    v8ff ka = *(const v8ff*)(kb + mc + hi * 16);
    v8ff kc = *(const v8ff*)(kb + mc + hi * 16 + 8);
    v16h A;
    #pragma unroll
    for (int j = 0; j < 16; ++j) {
      int K = hi * 16 + j;
      float kvj = (j < 8) ? ka[j] : kc[j - 8];
      float e = __builtin_amdgcn_exp2f(ql2 * kvj - rm2);
      e = ((mc + K) < SEQ) ? e : 0.f;
      A[j] = (_Float16)e;
    }
    v16h B = *(const v16h*)(vb + mc);
    acc = __builtin_amdgcn_wmma_f32_16x16x32_f16(false, A, false, B,
                                                 (short)0, acc, false, false);
  }

  // D[M=r+8*hi][h=lm] per lane; row 8 of Vpad gives Z in column h==8 (lanes 8 / 24)
  float w2h = (lm < 8) ? params[112 + lm] : 0.f;
  float b2v = params[34];
  #pragma unroll
  for (int r = 0; r < 8; ++r) {
    float Zlo = __builtin_amdgcn_readlane(acc[r], 8);    // rows 0-7 denominators
    float Zhi = __builtin_amdgcn_readlane(acc[r], 24);   // rows 8-15 denominators
    float Z = hi ? Zhi : Zlo;
    float gZ = gamma * __builtin_amdgcn_rcpf(Z);         // Z >= 1 (max element -> E=1)
    int n = n0 + r + hi * 8;
    float xv = (lm < 8 && n < SEQ) ? x[(b * 8 + lm) * SEQ + n] : 0.f;
    float val = w2h * (acc[r] * gZ + xv);                // fold residual + l2 projection
    val += __shfl_xor(val, 1, 32);
    val += __shfl_xor(val, 2, 32);
    val += __shfl_xor(val, 4, 32);
    val += __shfl_xor(val, 8, 32);
    if (lm == 0 && n < SEQ) ypre[b * SEQ + n] = val + b2v;
  }
}

__global__ void gen_k4_bn2(const float* __restrict__ ypre,
                           const float* __restrict__ g2, const float* __restrict__ be2,
                           float* __restrict__ out) {
  __shared__ float ss[256], ss2[256];
  __shared__ float sm, sv;
  int t = threadIdx.x;
  float s = 0.f, s2 = 0.f;
  for (int i = t; i < NTOT; i += 256) { float v = ypre[i]; s += v; s2 += v * v; }
  ss[t] = s; ss2[t] = s2; __syncthreads();
  for (int k = 128; k > 0; k >>= 1) {
    if (t < k) { ss[t] += ss[t + k]; ss2[t] += ss2[t + k]; }
    __syncthreads();
  }
  if (t == 0) { sm = ss[0] / (float)NTOT; sv = ss2[0] / (float)NTOT - sm * sm; }
  __syncthreads();
  float m = sm, sc = rsqrtf(sv + EPSBN) * g2[0], be = be2[0];
  for (int i = t; i < NTOT; i += 256)
    out[i] = fmaxf((ypre[i] - m) * sc + be, 0.f);
}

extern "C" void kernel_launch(void* const* d_in, const int* in_sizes, int n_in,
                              void* d_out, int out_size, void* d_ws, size_t ws_size,
                              hipStream_t stream) {
  const float* z    = (const float*)d_in[0];
  const float* w1   = (const float*)d_in[1];
  const float* b1   = (const float*)d_in[2];
  const float* g1   = (const float*)d_in[3];
  const float* be1  = (const float*)d_in[4];
  const float* wq   = (const float*)d_in[5];
  const float* bq   = (const float*)d_in[6];
  const float* wk   = (const float*)d_in[7];
  const float* bk   = (const float*)d_in[8];
  const float* wv   = (const float*)d_in[9];
  const float* bv   = (const float*)d_in[10];
  const float* gam  = (const float*)d_in[11];
  const float* w2   = (const float*)d_in[12];
  const float* b2   = (const float*)d_in[13];
  const float* g2   = (const float*)d_in[14];
  const float* be2  = (const float*)d_in[15];

  char* ws = (char*)d_ws;
  size_t off = 0;
  auto alloc = [&](size_t bytes) {
    size_t o = off;
    off = (off + bytes + 255) & ~(size_t)255;
    return o;
  };
  float*    params = (float*)   (ws + alloc(256 * sizeof(float)));
  float*    xbuf   = (float*)   (ws + alloc((size_t)BATCH * 8 * SEQ * sizeof(float)));
  float*    qbuf   = (float*)   (ws + alloc((size_t)BATCH * SEQ * sizeof(float)));
  float*    kbuf   = (float*)   (ws + alloc((size_t)BATCH * NPAD * sizeof(float)));
  _Float16* vpad   = (_Float16*)(ws + alloc((size_t)BATCH * 16 * NPAD * sizeof(_Float16)));
  float*    kstat  = (float*)   (ws + alloc((size_t)BATCH * 2 * sizeof(float)));
  float*    ypre   = (float*)   (ws + alloc((size_t)BATCH * SEQ * sizeof(float)));
  (void)ws_size; (void)in_sizes; (void)n_in; (void)out_size;

  gen_k1_stats<<<1, 256, 0, stream>>>(z, w1, b1, g1, be1, wq, bq, wk, bk,
                                      wv, bv, w2, b2, params);
  gen_k2_features<<<(BATCH * NPAD) / 256, 256, 0, stream>>>(z, params, xbuf, qbuf, kbuf, vpad);
  gen_k2b_kstat<<<BATCH, 256, 0, stream>>>(kbuf, kstat);
  gen_k3_attn<<<BATCH * TPB, 128, 0, stream>>>(qbuf, kbuf, vpad, xbuf, kstat,
                                               params, gam, ypre);
  gen_k4_bn2<<<1, 256, 0, stream>>>(ypre, g2, be2, (float*)d_out);
}
